// SwinTransformerBlock_24300924961163
// MI455X (gfx1250) — matrix operand order repaired and checked by reading, not compiled, not run on hardware
//
#include <hip/hip_runtime.h>
#include <hip/hip_bf16.h>
#include <math.h>

// ---------------------------------------------------------------------------
// SwinV2 block, CDNA5 (gfx1250), wave32, bf16 WMMA everywhere.
// B=32, H=W=56, WS=7, SHIFT=3, C=512, HEADS=16, hd=32, HID=2048
// N=49 tokens/window padded to 64 for 16x16 WMMA tiling. 2048 windows.
// ---------------------------------------------------------------------------

typedef __attribute__((ext_vector_type(16))) __bf16 v16bf;
typedef __attribute__((ext_vector_type(8)))  float  v8f;

__device__ __forceinline__ v8f wmma_bf16(v16bf a, v16bf b, v8f c) {
  // 8 args: (neg_a, A, neg_b, B, c_mod, C, reuse_a, reuse_b)
  return __builtin_amdgcn_wmma_f32_16x16x32_bf16(false, a, false, b,
                                                 (short)0, c, false, false);
}

// A-fragment (16x32 bf16): lane holds row M=lane&15; K = 16*half+{0..15}
// -> one contiguous 32B run.
__device__ __forceinline__ v16bf load_a(const __bf16* lo, const __bf16* hi) {
  v16bf f;
#pragma unroll
  for (int i = 0; i < 8; ++i) { f[i] = lo[i]; f[i + 8] = lo[i + 8]; }
  return f;
}

// B-fragment (32x16 bf16): lane holds col N=lane&15; K = 16*half+{0..15}
// contiguous along K in an N-major (row = N, contiguous K) source.
__device__ __forceinline__ v16bf load_b16(const __bf16* p) {
  v16bf f;
#pragma unroll
  for (int i = 0; i < 16; ++i) f[i] = p[i];
  return f;
}

// B-fragment from K-major storage (stride between consecutive K elements).
__device__ __forceinline__ v16bf load_b_strided(const __bf16* p, int stride) {
  v16bf f;
#pragma unroll
  for (int i = 0; i < 16; ++i) f[i] = p[i * stride];
  return f;
}

__device__ __forceinline__ v16bf zero_frag() {
  v16bf f;
#pragma unroll
  for (int i = 0; i < 16; ++i) f[i] = (__bf16)0.0f;
  return f;
}

// ---------------------------------------------------------------------------
// fp32 -> bf16 conversion (grid-stride)
// ---------------------------------------------------------------------------
__global__ void cvt_f32_bf16(const float* __restrict__ src,
                             __bf16* __restrict__ dst, int n) {
  for (int i = blockIdx.x * blockDim.x + threadIdx.x; i < n;
       i += gridDim.x * blockDim.x)
    dst[i] = (__bf16)src[i];
}

// ---------------------------------------------------------------------------
// Continuous relative position bias, stage 1: hid[169][512] = relu(tbl@W1^T+b1)
// ---------------------------------------------------------------------------
__global__ void cpb_hidden(const float* __restrict__ tbl,
                           const float* __restrict__ w1,
                           const float* __restrict__ b1,
                           float* __restrict__ hid) {
  int i = blockIdx.x * blockDim.x + threadIdx.x;
  if (i < 169 * 512) {
    int r = i >> 9, j = i & 511;
    float v = tbl[r * 2] * w1[j * 2] + tbl[r * 2 + 1] * w1[j * 2 + 1] + b1[j];
    hid[i] = v > 0.f ? v : 0.f;
  }
}

// stage 2: tab[169][16] = hid @ W2^T; biasTab[h][i][j]=16*sigmoid(tab[rel[i,j]][h])
__global__ void cpb_bias(const float* __restrict__ hid,
                         const float* __restrict__ w2,
                         const int* __restrict__ rel_index,
                         float* __restrict__ biasTab) {
  __shared__ float tab[169 * 16];
  for (int t = threadIdx.x; t < 169 * 16; t += blockDim.x) {
    int r = t >> 4, h = t & 15;
    const float* hr = hid + r * 512;
    const float* wr = w2 + h * 512;
    float s = 0.f;
    for (int k = 0; k < 512; ++k) s += hr[k] * wr[k];
    tab[t] = s;
  }
  __syncthreads();
  for (int t = threadIdx.x; t < 16 * 2401; t += blockDim.x) {
    int h = t / 2401, ij = t % 2401;
    float v = tab[rel_index[ij] * 16 + h];
    biasTab[t] = 16.f / (1.f + expf(-v));
  }
}

// ---------------------------------------------------------------------------
// Fused shifted-window attention: async gather -> QKV WMMA -> cosine-norm ->
// scores WMMA -> in-register softmax -> P@V WMMA -> attn_out (bf16)
// One block per window (2048 blocks), 256 threads (8 waves).
// LDS: X/P 64KB | Q 64KB | K 64KB | V 64KB  = 256KB
// ---------------------------------------------------------------------------
__global__ __launch_bounds__(256) void attn_kernel(
    const __bf16* __restrict__ xbf, const __bf16* __restrict__ wqkv,
    const float* __restrict__ q_bias, const float* __restrict__ v_bias,
    const float* __restrict__ logit_scale, const float* __restrict__ biasTab,
    const float* __restrict__ attn_mask, __bf16* __restrict__ attn_out) {
  extern __shared__ char smem[];
  __bf16* X  = (__bf16*)smem;             // 64x512 input (reused for P slabs)
  __bf16* Q  = (__bf16*)(smem + 65536);   // 64x512
  __bf16* Kl = (__bf16*)(smem + 131072);  // 64x512
  __bf16* V  = (__bf16*)(smem + 196608);  // 64x512

  const int w = blockIdx.x;
  const int b = w >> 6, wi = w & 63, wr = wi >> 3, wc = wi & 7;
  const int tid = threadIdx.x, lane = tid & 31, wv = tid >> 5;
  const int m = lane & 15, half = lane >> 4;

  // phase 1: cyclic-shift + window-partition gather. Real rows are 1KB
  // contiguous in pre-converted bf16 x -> DMA them straight into LDS with
  // per-lane GLOBAL_LOAD_ASYNC_TO_LDS_B128 (ASYNCcnt-tracked, no VGPR data).
  for (int idx = tid; idx < 15 * 512; idx += 256)  // zero pad rows 49..63
    X[49 * 512 + idx] = (__bf16)0.0f;
  {
    const unsigned dynbase = __builtin_amdgcn_groupstaticsize();  // X offset
    for (int t = wv; t < 49; t += 8) {
      int hh = (wr * 7 + t / 7 + 3) % 56;
      int ww = (wc * 7 + t % 7 + 3) % 56;
      const __bf16* src = xbf + (((size_t)b * 3136) + hh * 56 + ww) * 512;
      unsigned lds0 = dynbase + (unsigned)t * 1024u + (unsigned)lane * 16u;
      const __bf16* g0 = src + lane * 8;   // 16B per lane, 512B per wave
      const __bf16* g1 = g0 + 256;         // second 512B half of the row
      asm volatile("global_load_async_to_lds_b128 %0, %1, off"
                   :: "v"(lds0), "v"(g0) : "memory");
      asm volatile("global_load_async_to_lds_b128 %0, %1, off"
                   :: "v"(lds0 + 512u), "v"(g1) : "memory");
    }
    asm volatile("s_wait_asynccnt 0" ::: "memory");
  }
  __syncthreads();

  // phase 2: QKV GEMM  (64x512)@(512x1536)^T  -> Q/K/V in LDS
  for (int tile = wv; tile < 384; tile += 8) {
    int mt = tile & 3, nt = tile >> 2;  // nt in [0,96)
    v8f acc = {};
    for (int k = 0; k < 16; ++k) {
      const __bf16* ap = X + (mt * 16 + m) * 512 + k * 32 + 16 * half;
      const __bf16* bp = wqkv + (size_t)(nt * 16 + m) * 512 + k * 32 + 16 * half;
      __builtin_prefetch(bp + 32, 0, 0);
      acc = wmma_bf16(load_a(ap, ap + 16), load_b16(bp), acc);
    }
    // destination region is tile-uniform (16-wide N tile never straddles
    // the Q/K/V boundaries) -> scalar branch, no per-lane EXEC churn.
    __bf16* dst;
    float bias;
    int ocr;
    if (nt < 32) {
      dst = Q;  ocr = nt * 16 + m;         bias = q_bias[ocr];
    } else if (nt < 64) {
      dst = Kl; ocr = (nt - 32) * 16 + m;  bias = 0.f;
    } else {
      dst = V;  ocr = (nt - 64) * 16 + m;  bias = v_bias[ocr];
    }
#pragma unroll
    for (int r = 0; r < 8; ++r) {
      int row = mt * 16 + r + 8 * half;
      dst[row * 512 + ocr] = (__bf16)(acc[r] + bias);
    }
  }
  __syncthreads();

  // phase 3: cosine-attention row normalization; fold logit scale into q
  for (int task = tid; task < 2 * 49 * 16; task += 256) {
    int which = task & 1, rem = task >> 1;
    int t = rem % 49, h = rem / 49;
    __bf16* base = (which ? Kl : Q) + t * 512 + h * 32;
    float ss = 0.f;
    for (int d = 0; d < 32; ++d) { float f = (float)base[d]; ss += f * f; }
    float inv = 1.f / fmaxf(sqrtf(ss), 1e-12f);
    if (!which) inv *= expf(fminf(logit_scale[h], 4.6051702f));  // ln(100)
    for (int d = 0; d < 32; ++d) base[d] = (__bf16)((float)base[d] * inv);
  }
  __syncthreads();

  // phase 4: per-head attention; each wave owns heads {wv, wv+8}
  __bf16* Pw = X + wv * 4096;  // per-wave 64x64 bf16 slab (X is dead)
  for (int hi = 0; hi < 2; ++hi) {
    int h = wv + hi * 8;
    float sreg[4][4][8];
    // scores: S = qn @ kn^T, K=hd=32 -> single WMMA per 16x16 tile
#pragma unroll
    for (int mt = 0; mt < 4; ++mt)
#pragma unroll
      for (int nt = 0; nt < 4; ++nt) {
        const __bf16* ap = Q + (mt * 16 + m) * 512 + h * 32 + 16 * half;
        const __bf16* bp = Kl + (nt * 16 + m) * 512 + h * 32 + 16 * half;
        v8f acc = {};
        acc = wmma_bf16(load_a(ap, ap + 16), load_b16(bp), acc);
#pragma unroll
        for (int r = 0; r < 8; ++r) {
          int row = mt * 16 + r + 8 * half;
          int col = nt * 16 + m;
          float s = -1e30f;
          if (row < 49 && col < 49)
            s = acc[r] + biasTab[h * 2401 + row * 49 + col] +
                attn_mask[wi * 2401 + row * 49 + col];
          sreg[mt][nt][r] = s;
        }
      }
    // softmax fully in registers: row = r + 8*half (+16*mt), cols striped
    // across the 16 lanes of each half -> width-16 shuffle reductions.
#pragma unroll
    for (int mt = 0; mt < 4; ++mt)
#pragma unroll
      for (int r = 0; r < 8; ++r) {
        float mx = fmaxf(fmaxf(sreg[mt][0][r], sreg[mt][1][r]),
                         fmaxf(sreg[mt][2][r], sreg[mt][3][r]));
#pragma unroll
        for (int off = 8; off; off >>= 1)
          mx = fmaxf(mx, __shfl_xor(mx, off, 16));
        float sum = 0.f;
#pragma unroll
        for (int nt = 0; nt < 4; ++nt) {
          float p = expf(sreg[mt][nt][r] - mx);
          sreg[mt][nt][r] = p;
          sum += p;
        }
#pragma unroll
        for (int off = 8; off; off >>= 1) sum += __shfl_xor(sum, off, 16);
        float rinv = sum > 0.f ? 1.f / sum : 0.f;
#pragma unroll
        for (int nt = 0; nt < 4; ++nt) sreg[mt][nt][r] *= rinv;
      }
    // bounce P through LDS to re-fragment as WMMA-A
#pragma unroll
    for (int mt = 0; mt < 4; ++mt)
#pragma unroll
      for (int nt = 0; nt < 4; ++nt)
#pragma unroll
        for (int r = 0; r < 8; ++r) {
          int row = mt * 16 + r + 8 * half, col = nt * 16 + m;
          Pw[row * 64 + col] = (__bf16)sreg[mt][nt][r];
        }
    // O = P @ V : (64x64)@(64x32), K split in two 32-chunks
#pragma unroll
    for (int mt = 0; mt < 4; ++mt)
#pragma unroll
      for (int ntv = 0; ntv < 2; ++ntv) {
        v8f acc = {};
#pragma unroll
        for (int kt = 0; kt < 2; ++kt) {
          const __bf16* ap = Pw + (mt * 16 + m) * 64 + kt * 32 + 16 * half;
          const __bf16* bp =
              V + (size_t)(kt * 32 + 16 * half) * 512 + h * 32 + ntv * 16 + m;
          acc = wmma_bf16(load_a(ap, ap + 16), load_b_strided(bp, 512), acc);
        }
#pragma unroll
        for (int r = 0; r < 8; ++r) {
          int row = mt * 16 + r + 8 * half;
          if (row < 49)
            attn_out[((size_t)w * 49 + row) * 512 + h * 32 + ntv * 16 + m] =
                (__bf16)acc[r];
        }
      }
  }
}

// ---------------------------------------------------------------------------
// Proj GEMM + window-reverse/roll + post-norm LN + residual -> x1 (fp32)
// One block per window, LDS: 64x512 fp32 = 128KB
// ---------------------------------------------------------------------------
__global__ __launch_bounds__(256) void proj_kernel(
    const __bf16* __restrict__ attn_out, const __bf16* __restrict__ wproj,
    const float* __restrict__ proj_b, const float* __restrict__ x,
    const float* __restrict__ g1, const float* __restrict__ bb1,
    float* __restrict__ x1) {
  extern __shared__ char smem[];
  float* O = (float*)smem;  // 64x512
  const int w = blockIdx.x;
  const int b = w >> 6, wi = w & 63, wr = wi >> 3, wc = wi & 7;
  const int tid = threadIdx.x, lane = tid & 31, wv = tid >> 5;
  const int m = lane & 15, half = lane >> 4;

  for (int tile = wv; tile < 128; tile += 8) {
    int mt = tile & 3, nt = tile >> 2;
    v8f acc = {};
    for (int k = 0; k < 16; ++k) {
      int row = mt * 16 + m;
      v16bf a;
      if (row < 49) {
        const __bf16* ap =
            attn_out + ((size_t)w * 49 + row) * 512 + k * 32 + 16 * half;
        a = load_a(ap, ap + 16);
      } else {
        a = zero_frag();
      }
      const __bf16* bp = wproj + (size_t)(nt * 16 + m) * 512 + k * 32 + 16 * half;
      acc = wmma_bf16(a, load_b16(bp), acc);
    }
    int col = nt * 16 + m;
#pragma unroll
    for (int r = 0; r < 8; ++r) {
      int row = mt * 16 + r + 8 * half;
      O[row * 512 + col] = acc[r] + proj_b[col];
    }
  }
  __syncthreads();

  for (int t = tid; t < 49; t += 256) {
    const float* row = O + t * 512;
    float mu = 0.f;
    for (int c = 0; c < 512; ++c) mu += row[c];
    mu *= (1.f / 512.f);
    float var = 0.f;
    for (int c = 0; c < 512; ++c) { float d = row[c] - mu; var += d * d; }
    var *= (1.f / 512.f);
    float rs = rsqrtf(var + 1e-5f);
    int hh = (wr * 7 + t / 7 + 3) % 56;
    int ww = (wc * 7 + t % 7 + 3) % 56;
    size_t dst = (((size_t)b * 3136) + hh * 56 + ww) * 512;
    for (int c = 0; c < 512; ++c)
      x1[dst + c] = x[dst + c] + (row[c] - mu) * rs * g1[c] + bb1[c];
  }
}

// ---------------------------------------------------------------------------
// Fused MLP: fc1 WMMA + GELU -> hidden in LDS -> fc2 WMMA -> LN -> residual
// One block per 32 rows (3136 blocks). LDS: H 128KB | X 32KB / O 64KB = 192KB
// ---------------------------------------------------------------------------
__global__ __launch_bounds__(256) void mlp_kernel(
    const float* __restrict__ x1, const __bf16* __restrict__ wfc1,
    const float* __restrict__ fc1_b, const __bf16* __restrict__ wfc2,
    const float* __restrict__ fc2_b, const float* __restrict__ g2,
    const float* __restrict__ b2, float* __restrict__ out) {
  extern __shared__ char smem[];
  __bf16* Hs = (__bf16*)smem;             // 32x2048 bf16 (131072 B)
  __bf16* Xs = (__bf16*)(smem + 131072);  // 32x512 bf16  (32768 B)
  float* O   = (float*)(smem + 131072);   // 32x512 fp32, overlaps Xs (dead)

  const int tid = threadIdx.x, lane = tid & 31, wv = tid >> 5;
  const int m = lane & 15, half = lane >> 4;
  const size_t rowbase = (size_t)blockIdx.x * 32;

  for (int idx = tid; idx < 32 * 512; idx += 256) {
    int t = idx >> 9, c = idx & 511;
    Xs[idx] = (__bf16)x1[(rowbase + t) * 512 + c];
  }
  __syncthreads();

  // fc1: (32x512)@(512x2048)^T + GELU
  for (int tile = wv; tile < 256; tile += 8) {
    int mt = tile & 1, nt = tile >> 1;
    v8f acc = {};
    for (int k = 0; k < 16; ++k) {
      const __bf16* ap = Xs + (mt * 16 + m) * 512 + k * 32 + 16 * half;
      const __bf16* bp = wfc1 + (size_t)(nt * 16 + m) * 512 + k * 32 + 16 * half;
      __builtin_prefetch(bp + 32, 0, 0);
      acc = wmma_bf16(load_a(ap, ap + 16), load_b16(bp), acc);
    }
    int col = nt * 16 + m;
#pragma unroll
    for (int r = 0; r < 8; ++r) {
      int row = mt * 16 + r + 8 * half;
      float h = acc[r] + fc1_b[col];
      h = 0.5f * h * (1.f + erff(h * 0.70710678f));  // exact GELU
      Hs[row * 2048 + col] = (__bf16)h;
    }
  }
  __syncthreads();

  // fc2: (32x2048)@(2048x512)^T
  for (int tile = wv; tile < 64; tile += 8) {
    int mt = tile & 1, nt = tile >> 1;
    v8f acc = {};
    for (int k = 0; k < 64; ++k) {
      const __bf16* ap = Hs + (mt * 16 + m) * 2048 + k * 32 + 16 * half;
      const __bf16* bp = wfc2 + (size_t)(nt * 16 + m) * 2048 + k * 32 + 16 * half;
      acc = wmma_bf16(load_a(ap, ap + 16), load_b16(bp), acc);
    }
    int col = nt * 16 + m;
#pragma unroll
    for (int r = 0; r < 8; ++r) {
      int row = mt * 16 + r + 8 * half;
      O[row * 512 + col] = acc[r] + fc2_b[col];
    }
  }
  __syncthreads();

  for (int t = tid; t < 32; t += 256) {
    const float* row = O + t * 512;
    float mu = 0.f;
    for (int c = 0; c < 512; ++c) mu += row[c];
    mu *= (1.f / 512.f);
    float var = 0.f;
    for (int c = 0; c < 512; ++c) { float d = row[c] - mu; var += d * d; }
    var *= (1.f / 512.f);
    float rs = rsqrtf(var + 1e-5f);
    size_t base = (rowbase + t) * 512;
    for (int c = 0; c < 512; ++c)
      out[base + c] = x1[base + c] + (row[c] - mu) * rs * g2[c] + b2[c];
  }
}

// ---------------------------------------------------------------------------
// Host launcher. Workspace layout (bytes):
//   0            wqkv  bf16 (1536x512)         1,572,864
//   1,572,864    wproj bf16 (512x512)            524,288
//   2,097,152    wfc1  bf16 (2048x512)         2,097,152
//   4,194,304    wfc2  bf16 (512x2048)         2,097,152
//   6,291,456    biasTab fp32 (16x49x49)         153,664
//   6,445,312    cpbHid fp32 (169x512)           346,112
//   6,791,424    attn_out bf16 (2048x49x512) 102,760,448
//   109,551,872  x1 fp32 (100352x512)        205,520,896
//   315,072,768  xbf bf16 (100352x512)       102,760,448
// total ~418 MB
// ---------------------------------------------------------------------------
extern "C" void kernel_launch(void* const* d_in, const int* in_sizes, int n_in,
                              void* d_out, int out_size, void* d_ws,
                              size_t ws_size, hipStream_t stream) {
  (void)in_sizes; (void)n_in; (void)out_size; (void)ws_size;
  const float* x           = (const float*)d_in[0];
  const float* qkv_w       = (const float*)d_in[1];
  const float* q_bias      = (const float*)d_in[2];
  const float* v_bias      = (const float*)d_in[3];
  const float* logit_scale = (const float*)d_in[4];
  const float* cpb_w1      = (const float*)d_in[5];
  const float* cpb_b1      = (const float*)d_in[6];
  const float* cpb_w2      = (const float*)d_in[7];
  const float* proj_w      = (const float*)d_in[8];
  const float* proj_b      = (const float*)d_in[9];
  const float* norm1_g     = (const float*)d_in[10];
  const float* norm1_b     = (const float*)d_in[11];
  const float* norm2_g     = (const float*)d_in[12];
  const float* norm2_b     = (const float*)d_in[13];
  const float* fc1_w       = (const float*)d_in[14];
  const float* fc1_b       = (const float*)d_in[15];
  const float* fc2_w       = (const float*)d_in[16];
  const float* fc2_b       = (const float*)d_in[17];
  const float* rel_table   = (const float*)d_in[18];
  const int*   rel_index   = (const int*)d_in[19];
  const float* attn_mask   = (const float*)d_in[20];

  char* ws = (char*)d_ws;
  __bf16* wqkv    = (__bf16*)(ws + 0);
  __bf16* wproj   = (__bf16*)(ws + 1572864);
  __bf16* wfc1    = (__bf16*)(ws + 2097152);
  __bf16* wfc2    = (__bf16*)(ws + 4194304);
  float*  biasTab = (float*)(ws + 6291456);
  float*  cpbHid  = (float*)(ws + 6445312);
  __bf16* attnO   = (__bf16*)(ws + 6791424);
  float*  x1      = (float*)(ws + 109551872);
  __bf16* xbf     = (__bf16*)(ws + 315072768);

  hipFuncSetAttribute((const void*)attn_kernel,
                      hipFuncAttributeMaxDynamicSharedMemorySize, 262144);
  hipFuncSetAttribute((const void*)proj_kernel,
                      hipFuncAttributeMaxDynamicSharedMemorySize, 131072);
  hipFuncSetAttribute((const void*)mlp_kernel,
                      hipFuncAttributeMaxDynamicSharedMemorySize, 196608);

  cvt_f32_bf16<<<512, 256, 0, stream>>>(qkv_w, wqkv, 1536 * 512);
  cvt_f32_bf16<<<256, 256, 0, stream>>>(proj_w, wproj, 512 * 512);
  cvt_f32_bf16<<<512, 256, 0, stream>>>(fc1_w, wfc1, 2048 * 512);
  cvt_f32_bf16<<<512, 256, 0, stream>>>(fc2_w, wfc2, 512 * 2048);
  cvt_f32_bf16<<<2048, 256, 0, stream>>>(x, xbf, 100352 * 512);
  cpb_hidden<<<(169 * 512 + 255) / 256, 256, 0, stream>>>(rel_table, cpb_w1,
                                                          cpb_b1, cpbHid);
  cpb_bias<<<1, 256, 0, stream>>>(cpbHid, cpb_w2, rel_index, biasTab);

  attn_kernel<<<2048, 256, 262144, stream>>>(xbf, wqkv, q_bias, v_bias,
                                             logit_scale, biasTab, attn_mask,
                                             attnO);
  proj_kernel<<<2048, 256, 131072, stream>>>(attnO, wproj, proj_b, x, norm1_g,
                                             norm1_b, x1);
  mlp_kernel<<<3136, 256, 196608, stream>>>(x1, wfc1, fc1_b, wfc2, fc2_b,
                                            norm2_g, norm2_b, (float*)d_out);
}
